// Model_84653805404851
// MI455X (gfx1250) — matrix operand order repaired and checked by reading, not compiled, not run on hardware
//
#include <hip/hip_runtime.h>
#include <hip/hip_bf16.h>

typedef __attribute__((ext_vector_type(16))) _Float16 v16h;
typedef __attribute__((ext_vector_type(8)))  float    v8f;

#define T_SAMP   32768
#define CCH      128
#define NATOMS   1024
#define HPAD     256
#define TPAD     (T_SAMP + 2*HPAD)        // 33280 halves per channel row
#define XPLEN    33024                    // 64 + 32768 + 192
#define NE       ((size_t)NATOMS * (size_t)T_SAMP)
#define KSPARSE  64
#define NW       32768                    // twiddle period

// ---------------- workspace layout (bytes, 256-aligned) ----------------
static constexpr size_t OFF_HF16  = 0;                          // (128, TPAD) f16, padded activations
static constexpr size_t SZ_HF16   = (size_t)CCH * TPAD * 2;
static constexpr size_t OFF_BANDS = OFF_HF16 + SZ_HF16;         // 64512 f32 (all band signals)
static constexpr size_t SZ_BANDS  = 64512ull * 4;
static constexpr size_t OFF_SPEC  = OFF_BANDS + SZ_BANDS;       // 16385 complex f32
static constexpr size_t SZ_SPEC   = 131328;
static constexpr size_t OFF_STATE = OFF_SPEC + SZ_SPEC;         // radix/list/band-list state
static constexpr size_t SZ_STATE  = 8192;
static constexpr size_t ZBYTES    = OFF_STATE + SZ_STATE;       // zero-per-call region
static constexpr size_t OFF_TW    = ZBYTES;                     // 32768 (cos,sin) pairs
static constexpr size_t SZ_TW     = 262144;
static constexpr size_t OFF_XPAD  = OFF_TW + SZ_TW;             // padded x, f16
static constexpr size_t SZ_XPAD   = XPLEN * 2;
static constexpr size_t OFF_FBW   = OFF_XPAD + SZ_XPAD;         // filter bank f16 (128x128)
static constexpr size_t SZ_FBW    = 32768;
static constexpr size_t OFF_DILW  = OFF_FBW + SZ_FBW;           // (7,3,128,128) f16 repacked
static constexpr size_t SZ_DILW   = 688128;
static constexpr size_t OFF_PWW   = OFF_DILW + SZ_DILW;         // (7,128,128) f16
static constexpr size_t SZ_PWW    = 229376;
static constexpr size_t OFF_UPW   = OFF_PWW + SZ_PWW;           // (1024,128) f16
static constexpr size_t SZ_UPW    = 262144;
static constexpr size_t OFF_MSKW  = OFF_UPW + SZ_UPW;           // (1024,128) f16
static constexpr size_t SZ_MSKW   = 262144;
static constexpr size_t OFF_H32   = OFF_MSKW + SZ_MSKW;         // (128,32768) f32 residual stream
static constexpr size_t SZ_H32    = (size_t)CCH * T_SAMP * 4;
static constexpr size_t OFF_Y16   = OFF_H32 + SZ_H32;           // (128,32768) f16 dilconv output
static constexpr size_t SZ_Y16    = (size_t)CCH * T_SAMP * 2;
static constexpr size_t WS_NEED   = OFF_Y16 + SZ_Y16;           // ~34.2 MB

// state layout (int indices):
// [0]=prefix/threshold  [1]=kremain  [2]=listCnt  [3]=eqTicket
// [4..259]=hist  [260..323]=idx list  [324..387]=val list (float bits)
// [388..393]=bandCnt  [400 + b*192 + s*3 ...]=band entries {c, tb, dv}
__constant__ int d_bandOff[6] = {0, 1024, 3072, 7168, 15360, 31744};

// ---------------- WMMA helpers (wave32, 16x16x32 f16 -> f32) ----------------
__device__ __forceinline__ v16h load_b16(const _Float16* p, size_t ldb, int lane) {
  v16h b;
  const int hi = (lane & 16) ? 8 : 0;
  const int n = lane & 15;
#pragma unroll
  for (int i = 0; i < 16; ++i) {
    const int k = (i < 8) ? (hi + i) : (16 + hi + (i - 8));
    b[i] = p[(size_t)k * ldb + n];
  }
  return b;
}

// A 16x32 f16 fragment per ISA layout: lane m = lane&15;
// VGPR r holds K pair at ((r&4)<<2) + (lane>=16 ? 8:0) + ((r&3)<<1)
__device__ __forceinline__ v16h load_a_frag(const _Float16* Atile, int lda, int lane) {
  const int m  = lane & 15;
  const int hi = (lane & 16) ? 8 : 0;
  const _Float16* p = Atile + (size_t)m * lda;
  v16h a;
#pragma unroll
  for (int r = 0; r < 8; ++r) {
    const int kb = ((r & 4) << 2) + hi + ((r & 3) << 1);
    a[2*r]   = p[kb];
    a[2*r+1] = p[kb+1];
  }
  return a;
}

// 16x64 tile: acc[s] += A(16x128) * B(128x16) for 4 adjacent 16-col subtiles.
// A fragment loaded once per K-chunk and fed to 4 WMMAs; the 4 B rows are a
// contiguous 128B run per lane (merges into wide b128 loads).
__device__ __forceinline__ void gemm_k128_n4(const _Float16* Atile, int lda,
                                             const _Float16* Bbase, size_t ldb,
                                             int lane, v8f acc[4]) {
#pragma unroll
  for (int c0 = 0; c0 < 128; c0 += 32) {
    if (c0 + 32 < 128)
      __builtin_prefetch(Bbase + (size_t)(c0 + 32 + lane) * ldb, 0, 3); // global_prefetch_b8
    const v16h a = load_a_frag(Atile + c0, lda, lane);
    const _Float16* brow = Bbase + (size_t)c0 * ldb;
#pragma unroll
    for (int s = 0; s < 4; ++s) {
      v16h b = load_b16(brow + 16 * s, ldb, lane);
      acc[s] = __builtin_amdgcn_wmma_f32_16x16x32_f16(false, a, false, b, (short)0, acc[s],
                                                      false, false);
    }
  }
}

// ---------------- prep kernels ----------------
__global__ void k_zero4(float4* p, size_t n) {
  size_t i = (size_t)blockIdx.x * blockDim.x + threadIdx.x;
  size_t st = (size_t)gridDim.x * blockDim.x;
  float4 z; z.x = z.y = z.z = z.w = 0.f;
  for (; i < n; i += st) p[i] = z;
}

__global__ void k_cvt(_Float16* dst, const float* src, int n) {
  int i = blockIdx.x * blockDim.x + threadIdx.x;
  if (i < n) dst[i] = (_Float16)src[i];
}

__global__ void k_dil_repack(_Float16* dst, const float* src) { // 7*3*128*128
  int i = blockIdx.x * blockDim.x + threadIdx.x;
  if (i >= 7*3*128*128) return;
  int c = i & 127, o = (i >> 7) & 127, rem = i >> 14;
  int k = rem % 3, blk = rem / 3;
  dst[i] = (_Float16)src[(((size_t)blk*128 + o)*128 + c)*3 + k];
}

__global__ void k_xpad(_Float16* dst, const float* x) {
  int i = blockIdx.x * blockDim.x + threadIdx.x;
  if (i >= XPLEN) return;
  dst[i] = (i >= 64 && i < 64 + T_SAMP) ? (_Float16)x[i - 64] : (_Float16)0.f;
}

__global__ void k_tw(float* tw) {
  int n = blockIdx.x * blockDim.x + threadIdx.x;
  if (n >= NW) return;
  float s, c;
  sincosf(6.283185307179586f * (float)n / (float)NW, &s, &c);
  tw[2*n] = c; tw[2*n + 1] = s;
}

// ---------------- GEMM kernels (each wave owns a 16x64 output tile) ----------------
// filter-bank: h[o,t] = sum_k w[o,k] x[t+k-64]; B row k = xpad + n0 + k (ldb = 1)
__global__ __launch_bounds__(256) void k_fb_gemm(const _Float16* __restrict__ fbw,
                                                 const _Float16* __restrict__ xpad,
                                                 float* __restrict__ h32,
                                                 _Float16* __restrict__ hf16) {
  const int lane = threadIdx.x & 31, wave = threadIdx.x >> 5;
  const int n0 = blockIdx.x * 64, m0 = wave * 16;
  v8f acc[4] = {};
  gemm_k128_n4(fbw + (size_t)m0 * 128, 128, xpad + n0, 1, lane, acc);
  const int rsel = (lane >> 4) << 3;
#pragma unroll
  for (int s = 0; s < 4; ++s) {
    const int n = n0 + 16*s + (lane & 15);
#pragma unroll
    for (int g = 0; g < 8; ++g) {
      const int m = m0 + g + rsel;
      float v = acc[s][g];
      h32[(size_t)m * T_SAMP + n] = v;
      hf16[(size_t)m * TPAD + HPAD + n] = (_Float16)v;
    }
  }
}

// dilated conv K=3 as three shifted 128x128 matmuls (+ bias) -> y16
__global__ __launch_bounds__(256) void k_dil_gemm(const _Float16* __restrict__ dilw,
                                                  const float* __restrict__ dilb,
                                                  const _Float16* __restrict__ hf16,
                                                  _Float16* __restrict__ y16, int d) {
  const int lane = threadIdx.x & 31, wave = threadIdx.x >> 5;
  const int n0 = blockIdx.x * 64, m0 = wave * 16;
  v8f acc[4] = {};
#pragma unroll
  for (int k = 0; k < 3; ++k) {
    const int shift = (k - 1) * d;
    gemm_k128_n4(dilw + (size_t)k * 128 * 128 + (size_t)m0 * 128, 128,
                 hf16 + HPAD + n0 + shift, TPAD, lane, acc);
  }
  const int rsel = (lane >> 4) << 3;
#pragma unroll
  for (int s = 0; s < 4; ++s) {
    const int n = n0 + 16*s + (lane & 15);
#pragma unroll
    for (int g = 0; g < 8; ++g) {
      const int m = m0 + g + rsel;
      y16[(size_t)m * T_SAMP + n] = (_Float16)(acc[s][g] + dilb[m]);
    }
  }
}

// pointwise + bias + residual + leaky_relu(0.2) -> h32, hf16
__global__ __launch_bounds__(256) void k_pw_gemm(const _Float16* __restrict__ pww,
                                                 const float* __restrict__ pwb,
                                                 const _Float16* __restrict__ y16,
                                                 float* __restrict__ h32,
                                                 _Float16* __restrict__ hf16) {
  const int lane = threadIdx.x & 31, wave = threadIdx.x >> 5;
  const int n0 = blockIdx.x * 64, m0 = wave * 16;
  v8f acc[4] = {};
  gemm_k128_n4(pww + (size_t)m0 * 128, 128, y16 + n0, T_SAMP, lane, acc);
  const int rsel = (lane >> 4) << 3;
#pragma unroll
  for (int s = 0; s < 4; ++s) {
    const int n = n0 + 16*s + (lane & 15);
#pragma unroll
    for (int g = 0; g < 8; ++g) {
      const int m = m0 + g + rsel;
      float v = acc[s][g] + pwb[m] + h32[(size_t)m * T_SAMP + n];
      v = (v > 0.f) ? v : 0.2f * v;
      h32[(size_t)m * T_SAMP + n] = v;
      hf16[(size_t)m * TPAD + HPAD + n] = (_Float16)v;
    }
  }
}

// fused up & mask projections: e = (up_w@h + up_b) * tanh(mask_w@h + mask_b)
// B fragments are shared between the two weight matrices (8 WMMAs per 4 B loads).
__global__ __launch_bounds__(256) void k_upmask(const _Float16* __restrict__ upw,
                                                const _Float16* __restrict__ mskw,
                                                const float* __restrict__ upb,
                                                const float* __restrict__ mskb,
                                                const _Float16* __restrict__ hf16,
                                                float* __restrict__ e) {
  const int lane = threadIdx.x & 31, wave = threadIdx.x >> 5;
  const int n0 = blockIdx.x * 64, m0 = blockIdx.y * 128 + wave * 16;
  v8f au[4] = {}, am[4] = {};
#pragma unroll
  for (int c0 = 0; c0 < 128; c0 += 32) {
    const _Float16* brow = hf16 + HPAD + n0 + (size_t)c0 * TPAD;
    v16h b[4];
#pragma unroll
    for (int s = 0; s < 4; ++s) b[s] = load_b16(brow + 16 * s, TPAD, lane);
    const v16h a1 = load_a_frag(upw  + (size_t)m0 * 128 + c0, 128, lane);
#pragma unroll
    for (int s = 0; s < 4; ++s)
      au[s] = __builtin_amdgcn_wmma_f32_16x16x32_f16(false, a1, false, b[s], (short)0, au[s],
                                                     false, false);
    const v16h a2 = load_a_frag(mskw + (size_t)m0 * 128 + c0, 128, lane);
#pragma unroll
    for (int s = 0; s < 4; ++s)
      am[s] = __builtin_amdgcn_wmma_f32_16x16x32_f16(false, a2, false, b[s], (short)0, am[s],
                                                     false, false);
  }
  const int rsel = (lane >> 4) << 3;
#pragma unroll
  for (int s = 0; s < 4; ++s) {
    const int n = n0 + 16*s + (lane & 15);
#pragma unroll
    for (int g = 0; g < 8; ++g) {
      const int m = m0 + g + rsel;
      float u  = au[s][g] + upb[m];
      float mk = tanhf(am[s][g] + mskb[m]);
      e[(size_t)m * T_SAMP + n] = u * mk;
    }
  }
}

// ---------------- top-64 radix select + sparsify (float4-vectorized streams) ----------------
__device__ __forceinline__ unsigned fkey(float f) {
  unsigned u = __float_as_uint(f);
  return (u & 0x80000000u) ? ~u : (u | 0x80000000u);
}

__global__ __launch_bounds__(256) void k_hist(const float4* __restrict__ e4, int* state, int pass) {
  __shared__ int lh[256];
  lh[threadIdx.x] = 0;
  __syncthreads();
  const unsigned prefix = (unsigned)state[0];
  const unsigned msk = (pass == 3) ? 0u : (0xFFFFFFFFu << (8 * (pass + 1)));
  size_t i = (size_t)blockIdx.x * blockDim.x + threadIdx.x;
  const size_t st = (size_t)gridDim.x * blockDim.x;
  for (; i < NE / 4; i += st) {
    float4 v = e4[i];
    const float vv[4] = {v.x, v.y, v.z, v.w};
#pragma unroll
    for (int j = 0; j < 4; ++j) {
      unsigned u = fkey(vv[j]);
      if ((u & msk) == (prefix & msk))
        atomicAdd(&lh[(u >> (8 * pass)) & 255], 1);
    }
  }
  __syncthreads();
  if (lh[threadIdx.x]) atomicAdd(&state[4 + threadIdx.x], lh[threadIdx.x]);
}

__global__ void k_scan(int* state, int pass) {
  if (threadIdx.x != 0) return;
  int kr = (pass == 3) ? KSPARSE : state[1];
  long cum = 0; int sel = 0;
  for (int b = 255; b >= 0; --b) {
    int c = state[4 + b];
    if (cum + c >= kr) { sel = b; break; }
    cum += c;
  }
  state[1] = kr - (int)cum;
  state[0] = (int)(((unsigned)state[0]) | ((unsigned)sel << (8 * pass)));
  for (int b = 0; b < 256; ++b) state[4 + b] = 0;
}

__global__ __launch_bounds__(256) void k_sparsify(float4* __restrict__ e4, int* state) {
  const unsigned thr = (unsigned)state[0];
  const int kr = state[1];
  size_t i = (size_t)blockIdx.x * blockDim.x + threadIdx.x;
  const size_t st = (size_t)gridDim.x * blockDim.x;
  for (; i < NE / 4; i += st) {
    float4 v4 = e4[i];
    float vv[4] = {v4.x, v4.y, v4.z, v4.w};
#pragma unroll
    for (int j = 0; j < 4; ++j) {
      float v = vv[j];
      unsigned u = fkey(v);
      bool keep = false;
      if (u > thr) keep = true;
      else if (u == thr) keep = (atomicAdd(&state[3], 1) < kr);
      if (keep) {
        int p = atomicAdd(&state[2], 1);
        if (p < KSPARSE) { state[260 + p] = (int)(4*i + j); ((float*)state)[324 + p] = v; }
      } else {
        vv[j] = 0.f;
      }
    }
    v4.x = vv[0]; v4.y = vv[1]; v4.z = vv[2]; v4.w = vv[3];
    e4[i] = v4;
  }
}

// ---------------- sparse band synthesis ----------------
// per band: maxpool(win) of sparse e -> dedup per (atom, bin) with max(.,0) for win>1
__global__ void k_bandlists(int* state, const float* __restrict__ gains) {
  __shared__ int   sc[64], sT[64];
  __shared__ int   stb[6 * 64];
  __shared__ float sdv[6 * 64];
  const int cnt = min(state[2], KSPARSE);
  const int tid = threadIdx.x;          // 0..383
  const int b = tid >> 6, s = tid & 63;
  if (b == 0 && s < cnt) {
    int idx = state[260 + s];
    sc[s] = idx >> 15;
    sT[s] = idx & (T_SAMP - 1);
  }
  __syncthreads();
  if (s < cnt) {
    int tb = sT[s] >> (5 - b);          // win = 32 >> b
    float v = ((float*)state)[324 + s];
    float dv = (b < 5) ? fmaxf(v, 0.f) : v;   // maxpool vs zeros in window
    stb[b * 64 + s] = tb; sdv[b * 64 + s] = dv;
  }
  __syncthreads();
  if (s < cnt) {
    int tb = stb[b * 64 + s];
    float dv = sdv[b * 64 + s];
    bool win = (dv != 0.f);
    if (win) {
      for (int j = 0; j < cnt; ++j) {
        if (j == s || sc[j] != sc[s] || stb[b * 64 + j] != tb) continue;
        float dj = sdv[b * 64 + j];
        if (dj > dv || (dj == dv && j < s)) { win = false; break; }
      }
    }
    if (win) {
      int p = atomicAdd(&state[388 + b], 1);
      int base = 400 + b * 192 + p * 3;
      state[base] = sc[s];
      state[base + 1] = tb;
      ((float*)state)[base + 2] = dv * gains[b];
    }
  }
}

// scatter-add gain*v*atom at position tb (truncated linear conv_transpose)
__global__ __launch_bounds__(256) void k_synth(const int* state, float* __restrict__ bands,
                                               const float* a0, const float* a1,
                                               const float* a2, const float* a3,
                                               const float* a4, const float* a5) {
  const int b = blockIdx.z, slot = blockIdx.y;
  if (slot >= state[388 + b]) return;
  const int Kb = 256 << b;
  const int j = blockIdx.x * 256 + threadIdx.x;
  if (j >= Kb) return;
  const int base = 400 + b * 192 + slot * 3;
  const int c = state[base], tb = state[base + 1];
  const float dv = ((const float*)state)[base + 2];
  const float* at;
  switch (b) { case 0: at = a0; break; case 1: at = a1; break; case 2: at = a2; break;
               case 3: at = a3; break; case 4: at = a4; break; default: at = a5; }
  const int bs = 1024 << b;
  const int pos = tb + j;
  if (pos < bs)
    atomicAdd(&bands[d_bandOff[b] + pos], dv * at[(size_t)c * Kb + j]);
}

// ---------------- spectrum assembly (ortho rfft of each band) + inverse ----------------
__global__ __launch_bounds__(256) void k_dft(const float* __restrict__ bands,
                                             const float* __restrict__ tw,
                                             float* __restrict__ S) {
  const int cnt[6] = {513, 513, 1025, 2049, 4097, 8193};
  const int jst[6] = {0, 512, 1024, 2048, 4096, 8192};
  int rem = blockIdx.x, b = 0;
  while (b < 5 && rem >= cnt[b]) { rem -= cnt[b]; ++b; }
  const int j = jst[b] + rem;
  const int bs = 1024 << b;
  const unsigned step = ((unsigned)j * (unsigned)(32 >> b)) & (NW - 1);
  const float* band = bands + d_bandOff[b];
  float re = 0.f, im = 0.f;
  for (int t = threadIdx.x; t < bs; t += 256) {
    unsigned n = ((unsigned)t * step) & (NW - 1);
    float v = band[t];
    re += v * tw[2 * n];
    im -= v * tw[2 * n + 1];
  }
  __shared__ float sr[256], si[256];
  const int tid = threadIdx.x;
  sr[tid] = re; si[tid] = im;
  __syncthreads();
  for (int s = 128; s > 0; s >>= 1) {
    if (tid < s) { sr[tid] += sr[tid + s]; si[tid] += si[tid + s]; }
    __syncthreads();
  }
  if (tid == 0) {
    float nrm = rsqrtf((float)bs);
    atomicAdd(&S[2 * j],     sr[0] * nrm);
    atomicAdd(&S[2 * j + 1], si[0] * nrm);
  }
}

__global__ __launch_bounds__(256) void k_idft(const float* __restrict__ S,
                                              const float* __restrict__ tw,
                                              float* __restrict__ out) {
  const int t = blockIdx.x;
  float acc = 0.f;
  for (int j = threadIdx.x; j <= NW / 2; j += 256) {
    unsigned n = ((unsigned)j * (unsigned)t) & (NW - 1);
    float w = (j == 0 || j == NW / 2) ? 1.f : 2.f;
    acc += w * (S[2 * j] * tw[2 * n] - S[2 * j + 1] * tw[2 * n + 1]);
  }
  __shared__ float sr[256];
  const int tid = threadIdx.x;
  sr[tid] = acc;
  __syncthreads();
  for (int s = 128; s > 0; s >>= 1) {
    if (tid < s) sr[tid] += sr[tid + s];
    __syncthreads();
  }
  if (tid == 0) out[t] = sr[0] * rsqrtf((float)NW);
}

// ---------------- launcher ----------------
extern "C" void kernel_launch(void* const* d_in, const int* in_sizes, int n_in,
                              void* d_out, int out_size, void* d_ws, size_t ws_size,
                              hipStream_t stream) {
  (void)in_sizes; (void)n_in; (void)out_size;
  if (ws_size < WS_NEED) return;

  const float* x      = (const float*)d_in[0];
  const float* fb     = (const float*)d_in[1];
  const float* dil_w  = (const float*)d_in[2];
  const float* dil_b  = (const float*)d_in[3];
  const float* pw_w   = (const float*)d_in[4];
  const float* pw_b   = (const float*)d_in[5];
  const float* up_w   = (const float*)d_in[6];
  const float* up_b   = (const float*)d_in[7];
  const float* mask_w = (const float*)d_in[8];
  const float* mask_b = (const float*)d_in[9];
  const float* gains  = (const float*)d_in[10];
  const float* atoms[6];
  for (int i = 0; i < 6; ++i) atoms[i] = (const float*)d_in[11 + i];

  float* out = (float*)d_out;            // final (32768)
  float* e   = out + T_SAMP;             // e (1024*32768), written in place

  char* ws = (char*)d_ws;
  _Float16* hf16  = (_Float16*)(ws + OFF_HF16);
  float*    bands = (float*)(ws + OFF_BANDS);
  float*    S     = (float*)(ws + OFF_SPEC);
  int*      state = (int*)(ws + OFF_STATE);
  float*    tw    = (float*)(ws + OFF_TW);
  _Float16* xpad  = (_Float16*)(ws + OFF_XPAD);
  _Float16* fbw   = (_Float16*)(ws + OFF_FBW);
  _Float16* dilw  = (_Float16*)(ws + OFF_DILW);
  _Float16* pww   = (_Float16*)(ws + OFF_PWW);
  _Float16* upw   = (_Float16*)(ws + OFF_UPW);
  _Float16* mskw  = (_Float16*)(ws + OFF_MSKW);
  float*    h32   = (float*)(ws + OFF_H32);
  _Float16* y16   = (_Float16*)(ws + OFF_Y16);

  // prep: clear stateful region, convert weights to f16, build twiddles
  k_zero4<<<2048, 256, 0, stream>>>((float4*)ws, ZBYTES / 16);
  k_cvt<<<(16384 + 255) / 256, 256, 0, stream>>>(fbw, fb, 16384);
  k_dil_repack<<<(7*3*128*128 + 255) / 256, 256, 0, stream>>>(dilw, dil_w);
  k_cvt<<<(7*128*128 + 255) / 256, 256, 0, stream>>>(pww, pw_w, 7*128*128);
  k_cvt<<<(1024*128 + 255) / 256, 256, 0, stream>>>(upw, up_w, 1024*128);
  k_cvt<<<(1024*128 + 255) / 256, 256, 0, stream>>>(mskw, mask_w, 1024*128);
  k_xpad<<<(XPLEN + 255) / 256, 256, 0, stream>>>(xpad, x);
  k_tw<<<NW / 256, 256, 0, stream>>>(tw);

  // filter-bank analysis (WMMA im2col GEMM)
  k_fb_gemm<<<T_SAMP / 64, 256, 0, stream>>>(fbw, xpad, h32, hf16);

  // 7 dilated residual blocks (WMMA)
  const int DIL[7] = {1, 3, 9, 27, 81, 243, 1};
  for (int i = 0; i < 7; ++i) {
    k_dil_gemm<<<T_SAMP / 64, 256, 0, stream>>>(dilw + (size_t)i * 3 * 128 * 128,
                                                dil_b + i * 128, hf16, y16, DIL[i]);
    k_pw_gemm<<<T_SAMP / 64, 256, 0, stream>>>(pww + (size_t)i * 128 * 128,
                                               pw_b + i * 128, y16, h32, hf16);
  }

  // up/mask projections fused (WMMA), write product straight into e-output
  k_upmask<<<dim3(T_SAMP / 64, NATOMS / 128), 256, 0, stream>>>(upw, mskw, up_b, mask_b,
                                                                hf16, e);

  // top-64 radix select over 33.5M values, then in-place sparsify + compact
  for (int p = 3; p >= 0; --p) {
    k_hist<<<8192, 256, 0, stream>>>((const float4*)e, state, p);
    k_scan<<<1, 32, 0, stream>>>(state, p);
  }
  k_sparsify<<<8192, 256, 0, stream>>>((float4*)e, state);

  // sparse band synthesis -> spectrum assembly -> inverse DFT
  k_bandlists<<<1, 384, 0, stream>>>(state, gains);
  k_synth<<<dim3(32, 64, 6), 256, 0, stream>>>(state, bands, atoms[0], atoms[1], atoms[2],
                                               atoms[3], atoms[4], atoms[5]);
  k_dft<<<16390, 256, 0, stream>>>(bands, tw, S);
  k_idft<<<T_SAMP, 256, 0, stream>>>(S, tw, out);
}
